// LSSViewTransformer_81372450390073
// MI455X (gfx1250) — matrix operand order repaired and checked by reading, not compiled, not run on hardware
//
#include <hip/hip_runtime.h>

// ---------------- problem constants ----------------
constexpr int B_ = 2, N_ = 6, CAM_C = 256, OUT_C = 64;
constexpr int FH = 16, FW = 44;
constexpr int D_BINS = 59;                    // (60-1)/1
constexpr int O_TOT = D_BINS + OUT_C;         // 123
constexpr int O_PAD = 128;                    // padded out-channels for WMMA tiling
constexpr int PIX_PER_CAM = FH * FW;          // 704
constexpr int NCAM = B_ * N_;                 // 12
constexpr int M_PIX = NCAM * PIX_PER_CAM;     // 8448 (multiple of 16; cam stride 704 = 44*16)
constexpr int NX = 128, NY = 128;
constexpr int OUT_ELEMS = B_ * OUT_C * NY * NX; // 2,097,152 floats

// ---------------- workspace layout (bytes) ----------------
constexpr size_t WS_WB16  = 0;                                        // 128*256 bf16 = 64 KB
constexpr size_t WS_COMB  = 65536;                                    // 12*9 f32
constexpr size_t WS_DEPTH = 66560;                                    // M_PIX*59 f32
constexpr size_t WS_CTX   = WS_DEPTH + (size_t)M_PIX * D_BINS * 4;    // M_PIX*64 f32
// total ~4.1 MB

// ---------------- WMMA types ----------------
typedef __attribute__((ext_vector_type(16))) __bf16 v16bf;
typedef __attribute__((ext_vector_type(8)))  float  v8f;

union Frag { v16bf v; unsigned int u[8]; };
static_assert(sizeof(Frag) == 32, "frag size");

__device__ __forceinline__ unsigned short f2bf(float f) {
  unsigned int u = __float_as_uint(f);
  unsigned int r = u + 0x7FFFu + ((u >> 16) & 1u);   // round-to-nearest-even
  return (unsigned short)(r >> 16);
}

// ---------------- kernel 0: convert W to bf16 (padded) + comb = R * inv(K) ----------------
__global__ void prep_kernel(const float* __restrict__ w,        // [123][256]
                            const float* __restrict__ intrins,  // [12][3][3]
                            const float* __restrict__ rots,     // [12][3][3]
                            unsigned short* __restrict__ wb,    // [128][256] bf16
                            float* __restrict__ comb)           // [12][3][3]
{
  int t = threadIdx.x;
  if (blockIdx.x < 128) {
    int e = blockIdx.x * 256 + t;       // 32768 elements
    int o = e >> 8, k = e & 255;
    wb[e] = (o < O_TOT) ? f2bf(w[o * 256 + k]) : (unsigned short)0;
  } else if (t < NCAM) {
    const float* K = intrins + t * 9;
    const float* R = rots + t * 9;
    float a = K[0], b = K[1], c = K[2],
          d = K[3], e = K[4], f = K[5],
          g = K[6], h = K[7], i = K[8];
    float det = a * (e * i - f * h) - b * (d * i - f * g) + c * (d * h - e * g);
    float inv = 1.0f / det;
    float ki[9];
    ki[0] =  (e * i - f * h) * inv; ki[1] = -(b * i - c * h) * inv; ki[2] =  (b * f - c * e) * inv;
    ki[3] = -(d * i - f * g) * inv; ki[4] =  (a * i - c * g) * inv; ki[5] = -(a * f - c * d) * inv;
    ki[6] =  (d * h - e * g) * inv; ki[7] = -(a * h - b * g) * inv; ki[8] =  (a * e - b * d) * inv;
    float* cm = comb + t * 9;
    #pragma unroll
    for (int r2 = 0; r2 < 3; ++r2)
      #pragma unroll
      for (int c2 = 0; c2 < 3; ++c2)
        cm[r2 * 3 + c2] = R[r2 * 3 + 0] * ki[0 * 3 + c2] +
                          R[r2 * 3 + 1] * ki[1 * 3 + c2] +
                          R[r2 * 3 + 2] * ki[2 * 3 + c2];
  }
}

// ---------------- kernel 1: bf16 WMMA GEMM (1x1 conv) + fused softmax ----------------
// grid = 528 blocks (16 pixels each), 256 threads = 8 waves; wave w owns out-channels [16w,16w+16)
constexpr int AROW = 264; // padded LDS row stride in ushorts (bank-spread)

__global__ void __launch_bounds__(256)
feat_gemm_softmax_kernel(const float* __restrict__ x,            // [12][256][704]
                         const unsigned int* __restrict__ wb,    // bf16 pairs: [128 rows][128 uints]
                         const float* __restrict__ bias,         // [123]
                         float* __restrict__ depth_out,          // [M_PIX][59]
                         float* __restrict__ ctx_out)            // [M_PIX][64]
{
  __shared__ unsigned short Asm[16 * AROW];  // 16 pixels x 256 K (bf16)
  __shared__ float Fsm[16 * O_PAD];          // 16 pixels x 128 out-channels (f32)

  const int tid   = threadIdx.x;
  const int wave  = tid >> 5;
  const int lane  = tid & 31;
  const int mBase = blockIdx.x * 16;
  const int cam   = mBase / PIX_PER_CAM;     // tile never straddles cameras (704 % 16 == 0)
  const int pBase = mBase - cam * PIX_PER_CAM;

  // Stage A tile: convert f32 -> bf16 into LDS. x addr = (cam*256+c)*704 + p.
  for (int idx = tid; idx < 16 * CAM_C; idx += 256) {
    int m = idx & 15;
    int c = idx >> 4;
    float v = x[(size_t)(cam * CAM_C + c) * PIX_PER_CAM + pBase + m];
    Asm[m * AROW + c] = f2bf(v);
  }
  __syncthreads();

  const int half  = lane >> 4;   // 0/1 lane half per ISA VGPR layout
  const int mn    = lane & 15;   // A: row m, B: col n
  const int oBase = wave * 16;

  v8f acc = {};
  const unsigned int* arow = (const unsigned int*)Asm + ((mn * AROW) >> 1);
  #pragma unroll
  for (int kb = 0; kb < CAM_C; kb += 32) {
    Frag a, b;
    #pragma unroll
    for (int v = 0; v < 8; ++v) {
      // 16-bit A 16x32 layout: K = (v/4)*16 + half*8 + (v%4)*2  (pairs contiguous)
      int ka = kb + ((v >> 2) << 4) + (half << 3) + ((v & 3) << 1);
      a.u[v] = arow[ka >> 1];
      // 16-bit B 32x16 layout: same K order as A: K = (v/4)*16 + half*8 + (v%4)*2
      int kB = kb + ((v >> 2) << 4) + (half << 3) + ((v & 3) << 1);
      b.u[v] = wb[((oBase + mn) << 7) + (kB >> 1)];
    }
    acc = __builtin_amdgcn_wmma_f32_16x16x32_bf16(false, a.v, false, b.v,
                                                  (short)0, acc, false, false);
  }

  // D layout: VGPR r -> M = r + half*8 ; N = lane%16. Add bias, park in LDS.
  {
    int o = oBase + mn;
    float bval = (o < O_TOT) ? bias[o] : 0.0f;
    #pragma unroll
    for (int r = 0; r < 8; ++r) {
      int m = r + (half << 3);
      Fsm[m * O_PAD + o] = acc[r] + bval;
    }
  }
  __syncthreads();

  // Softmax over channels [0,59): one thread per pixel (trivial cost).
  if (tid < 16) {
    const float* row = &Fsm[tid * O_PAD];
    float mx = row[0];
    for (int d = 1; d < D_BINS; ++d) mx = fmaxf(mx, row[d]);
    float s = 0.0f;
    for (int d = 0; d < D_BINS; ++d) s += __expf(row[d] - mx);
    float invs = 1.0f / s;
    float* dout = depth_out + (size_t)(mBase + tid) * D_BINS;
    for (int d = 0; d < D_BINS; ++d) dout[d] = __expf(row[d] - mx) * invs;
  }
  // Context channels [59,123) -> ctx buffer.
  for (int idx = tid; idx < 16 * OUT_C; idx += 256) {
    int m = idx >> 6;
    int c = idx & 63;
    ctx_out[(size_t)(mBase + m) * OUT_C + c] = Fsm[m * O_PAD + D_BINS + c];
  }
}

// ---------------- kernel 2: zero the BEV output ----------------
__global__ void zero_kernel(float* __restrict__ out, int n) {
  int i = blockIdx.x * blockDim.x + threadIdx.x;
  int stride = gridDim.x * blockDim.x;
  for (; i < n; i += stride) out[i] = 0.0f;
}

// ---------------- kernel 3: geometry + outer-product voxel scatter ----------------
// one block per pixel; LDS-staged depth/coords/ctx; 59x64 atomic adds swept by the block
__global__ void __launch_bounds__(256)
splat_kernel(const float* __restrict__ comb,   // [12][9]
             const float* __restrict__ trans,  // [12][3]
             const float* __restrict__ depth,  // [M_PIX][59]
             const float* __restrict__ ctx,    // [M_PIX][64]
             float* __restrict__ out)          // [2][64][128][128]
{
  __shared__ float dsh[D_BINS];
  __shared__ int   cxy[D_BINS];     // (cy<<8)|cx, or -1 when culled
  __shared__ float ctxsh[OUT_C];

  const int m   = blockIdx.x;
  const int t   = threadIdx.x;
  const int cam = m / PIX_PER_CAM;
  const int b   = cam / N_;
  const int p   = m - cam * PIX_PER_CAM;
  const int h   = p / FW;
  const int w   = p - h * FW;

  const float xs = (float)w * (703.0f / 43.0f);  // linspace(0, IMG_W-1, FW)
  const float ys = (float)h * 17.0f;             // linspace(0, IMG_H-1, FH)

  if (t < D_BINS) {
    const float* C = comb + cam * 9;
    const float* T = trans + cam * 3;
    float dv = 1.0f + (float)t;                  // depth bin center
    float px = xs * dv, py = ys * dv, pz = dv;
    float gx = C[0] * px + C[1] * py + C[2] * pz + T[0];
    float gy = C[3] * px + C[4] * py + C[5] * pz + T[1];
    float gz = C[6] * px + C[7] * py + C[8] * pz + T[2];
    int cx = (int)floorf((gx + 51.2f) / 0.8f);
    int cy = (int)floorf((gy + 51.2f) / 0.8f);
    int cz = (int)floorf((gz + 10.0f) / 20.0f);
    bool kept = (cx >= 0) && (cx < NX) && (cy >= 0) && (cy < NY) && (cz == 0);
    cxy[t] = kept ? ((cy << 8) | cx) : -1;
    dsh[t] = depth[(size_t)m * D_BINS + t];
  }
  if (t >= 64 && t < 64 + OUT_C) {
    ctxsh[t - 64] = ctx[(size_t)m * OUT_C + (t - 64)];
  }
  __syncthreads();

  for (int i = t; i < D_BINS * OUT_C; i += 256) {
    int d = i >> 6;
    int c = i & 63;
    int pk = cxy[d];
    if (pk >= 0) {
      int cx = pk & 255, cy = pk >> 8;
      atomicAdd(&out[(((size_t)b * OUT_C + c) * NY + cy) * NX + cx], ctxsh[c] * dsh[d]);
    }
  }
}

// ---------------- launcher ----------------
extern "C" void kernel_launch(void* const* d_in, const int* in_sizes, int n_in,
                              void* d_out, int out_size, void* d_ws, size_t ws_size,
                              hipStream_t stream) {
  (void)in_sizes; (void)n_in; (void)out_size; (void)ws_size;

  const float* x       = (const float*)d_in[0];
  const float* rots    = (const float*)d_in[1];
  const float* trans   = (const float*)d_in[2];
  const float* intrins = (const float*)d_in[3];
  const float* depth_w = (const float*)d_in[4];
  const float* depth_b = (const float*)d_in[5];
  float* out = (float*)d_out;

  char* ws = (char*)d_ws;
  unsigned short* wb16 = (unsigned short*)(ws + WS_WB16);
  float* comb  = (float*)(ws + WS_COMB);
  float* depth = (float*)(ws + WS_DEPTH);
  float* ctx   = (float*)(ws + WS_CTX);

  prep_kernel<<<129, 256, 0, stream>>>(depth_w, intrins, rots, wb16, comb);
  feat_gemm_softmax_kernel<<<M_PIX / 16, 256, 0, stream>>>(
      x, (const unsigned int*)wb16, depth_b, depth, ctx);
  zero_kernel<<<1024, 256, 0, stream>>>(out, OUT_ELEMS);
  splat_kernel<<<M_PIX, 256, 0, stream>>>(comb, trans, depth, ctx, out);
}
